// LSTMModel_46737834115351
// MI455X (gfx1250) — compile-verified, matrix operand order checked by reading, not run on hardware
//
#include <hip/hip_runtime.h>
#include <hip/hip_bf16.h>

#define V_DIM 50257
#define D_DIM 512
#define T_DIM 1024

typedef __attribute__((ext_vector_type(16))) __bf16 v16bf;
typedef __attribute__((ext_vector_type(8)))  float  v8f;

// ---------- helpers ----------
static __device__ __forceinline__ unsigned short f2bf(float x) {
    unsigned u = __builtin_bit_cast(unsigned, x);
    u += 0x7FFFu + ((u >> 16) & 1u);          // round-to-nearest-even
    return (unsigned short)(u >> 16);
}
static __device__ __forceinline__ float bf2f(unsigned short h) {
    unsigned u = ((unsigned)h) << 16;
    return __builtin_bit_cast(float, u);
}

// ---------- kernel 0: interleave+convert recurrent weights to bf16 ----------
// WTb[(d*4+gate)*512 + k] = bf16(W_gate[d*512 + k]); gate order i,f,o,g
__global__ void prep_wh_kernel(const float* __restrict__ W_hi,
                               const float* __restrict__ W_hf,
                               const float* __restrict__ W_ho,
                               const float* __restrict__ W_hg,
                               unsigned short* __restrict__ WTb) {
    int idx = blockIdx.x * 256 + threadIdx.x;          // 0 .. 4*512*512-1
    if (idx >= 4 * D_DIM * D_DIM) return;
    int k      = idx & (D_DIM - 1);
    int rowout = idx >> 9;                             // d*4+gate
    int gate   = rowout & 3;
    int d      = rowout >> 2;
    const float* src = (gate == 0) ? W_hi : (gate == 1) ? W_hf
                     : (gate == 2) ? W_ho : W_hg;
    WTb[idx] = f2bf(src[d * D_DIM + k]);
}

// ---------- kernel 0b: convert W_out to bf16 ----------
__global__ void prep_wout_kernel(const float* __restrict__ W_out,
                                 unsigned short* __restrict__ Wb) {
    int idx = blockIdx.x * 256 + threadIdx.x;
    if (idx >= V_DIM * D_DIM) return;
    Wb[idx] = f2bf(W_out[idx]);
}

// ---------- kernel 1: cooperative sequential LSTM (16 blocks x 128 threads) ----------
// Block b owns dims [b*32, b*32+32); thread t = d_local*4 + gate.
// Recurrent weights live in LDS (bf16, padded rows), staged via
// GLOBAL_LOAD_ASYNC_TO_LDS_B64 (ASYNCcnt path). h double-buffered in ws.
#define REC_BLOCKS   16
#define REC_THREADS  128
#define WROW_U32     256                 // 512 bf16 per (dim,gate) row
#define WROW_PAD     258                 // 1032 B: 8B-aligned, conflict-free (2t mod 64)
#define SMEM_BYTES   (REC_THREADS * WROW_PAD * 4 + D_DIM * 4 + REC_THREADS * 4 + 32 * 4)

__global__ __launch_bounds__(REC_THREADS, 1)
void lstm_rec_kernel(const int* __restrict__ idx,
                     const float* __restrict__ W_ii, const float* __restrict__ W_if,
                     const float* __restrict__ W_io, const float* __restrict__ W_ig,
                     const float* __restrict__ b_i,  const float* __restrict__ b_f,
                     const float* __restrict__ b_o,  const float* __restrict__ b_g,
                     const unsigned int* __restrict__ WTb_u32,   // bf16 pairs
                     float* __restrict__ hbuf,                   // [2][512], zeroed
                     unsigned int* __restrict__ bar,             // zeroed
                     unsigned short* __restrict__ Hb,            // [T][512] bf16
                     float* __restrict__ outTail) {              // h then c
    extern __shared__ unsigned char smem[];
    unsigned int* ldsW  = (unsigned int*)smem;
    float* h_sh  = (float*)(smem + REC_THREADS * WROW_PAD * 4);
    float* act_sh = h_sh + D_DIM;
    float* c_sh   = act_sh + REC_THREADS;

    const int tid  = threadIdx.x;
    const int blk  = blockIdx.x;
    const int gate = tid & 3;
    const int dloc = tid >> 2;                 // 0..31
    const int gdim = blk * 32 + dloc;          // global hidden dim

    // Stage this thread's 1KB weight row into LDS via async-to-LDS (no VGPR bounce).
    // Each b64 copy moves 8 bytes from global memory straight into LDS, tracked by ASYNCcnt.
    {
        unsigned lbase = (unsigned)(size_t)(ldsW + (size_t)tid * WROW_PAD);
        unsigned long long gbase =
            (unsigned long long)(size_t)(WTb_u32 + (size_t)(blk * REC_THREADS + tid) * WROW_U32);
        #pragma unroll 8
        for (int j = 0; j < 128; ++j) {                   // 128 x 8B = 1KB per lane
            unsigned laddr = lbase + (unsigned)(j * 8);
            unsigned long long ga = gbase + (unsigned long long)(j * 8);
            asm volatile("global_load_async_to_lds_b64 %0, %1, off"
                         :: "v"(laddr), "v"(ga) : "memory");
        }
        asm volatile("s_wait_asynccnt 0x0" ::: "memory");
    }
    if (tid < 32) c_sh[tid] = 0.0f;

    const float* embW = (gate == 0) ? W_ii : (gate == 1) ? W_if
                      : (gate == 2) ? W_io : W_ig;
    const float bval  = (gate == 0) ? b_i[gdim] : (gate == 1) ? b_f[gdim]
                      : (gate == 2) ? b_o[gdim] : b_g[gdim];
    const unsigned int* wrow = ldsW + (size_t)tid * WROW_PAD;

    __syncthreads();

    for (int t = 0; t < T_DIM; ++t) {
        const int tok = idx[t];
        const float* hcur = hbuf + ((t & 1) * D_DIM);
        float*       hnxt = hbuf + (((t + 1) & 1) * D_DIM);

        // stage h_t into LDS
        for (int i = tid; i < D_DIM; i += REC_THREADS) h_sh[i] = hcur[i];
        __syncthreads();

        float acc = embW[(size_t)gdim * V_DIM + tok] + bval;
        #pragma unroll 8
        for (int j = 0; j < WROW_U32; ++j) {
            unsigned int w = wrow[j];
            acc += bf2f((unsigned short)(w & 0xFFFFu)) * h_sh[2 * j]
                 + bf2f((unsigned short)(w >> 16))     * h_sh[2 * j + 1];
        }
        act_sh[tid] = (gate < 3) ? (1.0f / (1.0f + expf(-acc))) : tanhf(acc);
        __syncthreads();

        if (tid < 32) {
            float gi = act_sh[4 * tid + 0];
            float gf = act_sh[4 * tid + 1];
            float go = act_sh[4 * tid + 2];
            float gg = act_sh[4 * tid + 3];
            float c  = gf * c_sh[tid] + gi * gg;
            c_sh[tid] = c;
            float h  = go * tanhf(c);
            int gd   = blk * 32 + tid;
            hnxt[gd] = h;
            Hb[(size_t)t * D_DIM + gd] = f2bf(h);
            if (t == T_DIM - 1) {
                outTail[gd]          = h;
                outTail[D_DIM + gd]  = c;
            }
        }
        // cross-block barrier: monotonic counter, target 16*(t+1)
        __threadfence();
        __syncthreads();
        if (tid == 0) {
            atomicAdd(bar, 1u);
            unsigned target = (unsigned)(REC_BLOCKS * (t + 1));
            volatile unsigned* vb = (volatile unsigned*)bar;
            while (*vb < target) __builtin_amdgcn_s_sleep(2);
        }
        __syncthreads();
        __threadfence();
    }
}

// ---------- kernel 2: logits = Hb (TxD, bf16) @ Wb^T (VxD, bf16) + b, WMMA ----------
// 256 threads = 8 waves; block tile M=64, N=32; one 16x16 tile per wave.
union AFrag { uint4 q[2]; v16bf v; };

__global__ __launch_bounds__(256, 2)
void gemm_wmma_kernel(const unsigned short* __restrict__ Hb,
                      const unsigned short* __restrict__ Wb,
                      const float* __restrict__ b_out,
                      float* __restrict__ logits) {
    const int tid  = threadIdx.x;
    const int wave = tid >> 5;
    const int lane = tid & 31;
    const int m_tile = wave & 3;           // 0..3
    const int n_tile = wave >> 2;          // 0..1

    const int row = blockIdx.y * 64 + m_tile * 16 + (lane & 15);     // < 1024
    const int col = blockIdx.x * 32 + n_tile * 16 + (lane & 15);     // may exceed V
    const int colc = (col < V_DIM) ? col : (V_DIM - 1);
    const int k0 = (lane < 16) ? 0 : 8;    // 16-bit A/B fragment K-origin per half-wave

    const unsigned short* aptr = Hb + (size_t)row  * D_DIM;
    const unsigned short* bptr = Wb + (size_t)colc * D_DIM;
    const float bias = (col < V_DIM) ? b_out[col] : 0.0f;

    v8f acc = {};
    #pragma unroll
    for (int kk = 0; kk < D_DIM; kk += 32) {
        __builtin_prefetch(aptr + kk + 64, 0, 1);
        __builtin_prefetch(bptr + kk + 64, 0, 1);
        AFrag fa, fb;
        fa.q[0] = *reinterpret_cast<const uint4*>(aptr + kk + k0);
        fa.q[1] = *reinterpret_cast<const uint4*>(aptr + kk + k0 + 16);
        fb.q[0] = *reinterpret_cast<const uint4*>(bptr + kk + k0);
        fb.q[1] = *reinterpret_cast<const uint4*>(bptr + kk + k0 + 16);
        acc = __builtin_amdgcn_wmma_f32_16x16x32_bf16(
                false, fa.v, false, fb.v, (short)0, acc, false, false);
    }

    // C/D layout: VGPR r -> M = r (+8 for lanes>=16), N = lane%16
    const int m_base = blockIdx.y * 64 + m_tile * 16 + ((lane < 16) ? 0 : 8);
    if (col < V_DIM) {
        #pragma unroll
        for (int r = 0; r < 8; ++r) {
            logits[(size_t)(m_base + r) * V_DIM + col] = acc[r] + bias;
        }
    }
}

// ---------- kernel 3: in-place log_softmax over each V-row ----------
__global__ __launch_bounds__(256)
void log_softmax_kernel(float* __restrict__ logits) {
    __shared__ float red[256];
    const int tid = threadIdx.x;
    float* p = logits + (size_t)blockIdx.x * V_DIM;

    float mx = -3.402823466e38f;
    for (int i = tid; i < V_DIM; i += 256) mx = fmaxf(mx, p[i]);
    red[tid] = mx; __syncthreads();
    for (int s = 128; s > 0; s >>= 1) {
        if (tid < s) red[tid] = fmaxf(red[tid], red[tid + s]);
        __syncthreads();
    }
    mx = red[0]; __syncthreads();

    float sum = 0.0f;
    for (int i = tid; i < V_DIM; i += 256) sum += expf(p[i] - mx);
    red[tid] = sum; __syncthreads();
    for (int s = 128; s > 0; s >>= 1) {
        if (tid < s) red[tid] += red[tid + s];
        __syncthreads();
    }
    float lse = mx + logf(red[0]);
    __syncthreads();
    for (int i = tid; i < V_DIM; i += 256) p[i] = p[i] - lse;
}

// ---------- host ----------
extern "C" void kernel_launch(void* const* d_in, const int* in_sizes, int n_in,
                              void* d_out, int out_size, void* d_ws, size_t ws_size,
                              hipStream_t stream) {
    (void)in_sizes; (void)n_in; (void)out_size; (void)ws_size;

    const int*   idx   = (const int*)  d_in[0];
    const float* W_ii  = (const float*)d_in[1];
    const float* W_if  = (const float*)d_in[2];
    const float* W_io  = (const float*)d_in[3];
    const float* W_ig  = (const float*)d_in[4];
    const float* W_hi  = (const float*)d_in[5];
    const float* W_hf  = (const float*)d_in[6];
    const float* W_ho  = (const float*)d_in[7];
    const float* W_hg  = (const float*)d_in[8];
    const float* b_i   = (const float*)d_in[9];
    const float* b_f   = (const float*)d_in[10];
    const float* b_o   = (const float*)d_in[11];
    const float* b_g   = (const float*)d_in[12];
    const float* W_out = (const float*)d_in[13];
    const float* b_out = (const float*)d_in[14];

    char* ws = (char*)d_ws;
    unsigned int*   bar  = (unsigned int*)(ws + 0);                  // 256 B
    float*          hbuf = (float*)(ws + 256);                       // 4 KB
    unsigned short* WTb  = (unsigned short*)(ws + 8192);             // 2 MB
    unsigned short* Hb   = (unsigned short*)(ws + 8192 + 2097152);   // 1 MB
    unsigned short* Wb   = (unsigned short*)(ws + 8192 + 2097152 + 1048576); // ~49 MB

    float* logits = (float*)d_out;
    float* tail   = logits + (size_t)T_DIM * V_DIM;

    // zero barrier + h double-buffer (capture-legal)
    hipMemsetAsync(ws, 0, 8192, stream);

    prep_wh_kernel<<<(4 * D_DIM * D_DIM) / 256, 256, 0, stream>>>(W_hi, W_hf, W_ho, W_hg, WTb);
    prep_wout_kernel<<<(V_DIM * D_DIM + 255) / 256, 256, 0, stream>>>(W_out, Wb);

    (void)hipFuncSetAttribute(reinterpret_cast<const void*>(lstm_rec_kernel),
                              hipFuncAttributeMaxDynamicSharedMemorySize, SMEM_BYTES);
    lstm_rec_kernel<<<REC_BLOCKS, REC_THREADS, SMEM_BYTES, stream>>>(
        idx, W_ii, W_if, W_io, W_ig, b_i, b_f, b_o, b_g,
        (const unsigned int*)WTb, hbuf, bar, Hb, tail);

    dim3 gg((V_DIM + 31) / 32, T_DIM / 64);   // (1571, 16)
    gemm_wmma_kernel<<<gg, 256, 0, stream>>>(Hb, Wb, b_out, logits);

    log_softmax_kernel<<<T_DIM, 256, 0, stream>>>(logits);
}
